// BDeformConv_45603962749564
// MI455X (gfx1250) — compile-verified, matrix operand order checked by reading, not run on hardware
//
#include <hip/hip_runtime.h>

// ---------------- problem constants ----------------
#define BATCH   2
#define CIN     64
#define COUT    64
#define H       192
#define W       192
#define HW      (H*W)            // 36864
#define NPIX    (BATCH*HW)       // 73728
#define KTAPS   9
#define KDIM    (KTAPS*CIN)      // 576
#define A_S     1.25f
#define B_S     1.75f

// ---------------- vector typedefs ----------------
typedef __attribute__((ext_vector_type(16))) __bf16  v16bf;
typedef __attribute__((ext_vector_type(8)))  float   v8f;
typedef __attribute__((ext_vector_type(4)))  unsigned int u32x4;
typedef __attribute__((ext_vector_type(4)))  float   f32x4;
typedef __attribute__((ext_vector_type(2)))  float   f32x2;

__device__ __forceinline__ unsigned short f2bf(float f) {
    return __builtin_bit_cast(unsigned short, (__bf16)f);   // native v_cvt
}

// ---------------- ws layout (bytes) ----------------
// [0)       : offsets (NPIX * 9 * 2 f32)            =  5,308,416
// [OFF_XF)  : x in NHWC f32 (NPIX * 64 * 4B)        = 18,874,368
// [OFF_WB)  : w_main bf16 B-layout (576*64*2B)      =     73,728
#define OFF_XF  (5308416u)
#define OFF_WB  (5308416u + 18874368u)

// ============================================================
// Kernel 1: w_main (64o,64c,3,3) f32 -> bf16 B-matrix blocks
// layout: [tap k][kblock kb][ntile nt][n 16][kk 32]  (bf16)
// ============================================================
__global__ void prep_wmain_kernel(const float* __restrict__ wmain,
                                  unsigned short* __restrict__ wb) {
    int t = blockIdx.x * blockDim.x + threadIdx.x;      // 576*64 threads
    if (t >= KDIM * COUT) return;
    int o  = t % COUT;
    int kk = t / COUT;          // kk = k*64 + c (tap-major K)
    int k  = kk / CIN;
    int c  = kk % CIN;
    int kb = c >> 5;
    int cl = c & 31;
    int nt = o >> 4;
    int nl = o & 15;
    float v = wmain[(o * CIN + c) * KTAPS + k];
    wb[(((k * 2 + kb) * 4 + nt) * 16 + nl) * 32 + cl] = f2bf(v);
}

// ============================================================
// Kernel 2: x NCHW -> NHWC (f32, stays in the 192MB L2)
// ============================================================
__global__ void convert_x_kernel(const float* __restrict__ x,
                                 float* __restrict__ xf) {
    int id = blockIdx.x * blockDim.x + threadIdx.x;     // NPIX*64 threads
    if (id >= NPIX * CIN) return;
    int c = id & 63;
    int p = id >> 6;
    int b = p / HW;
    int hw = p - b * HW;
    xf[id] = x[((long)(b * CIN + c)) * HW + hw];
}

// ============================================================
// Kernel 3: rotation / stretch / whole convs -> (py,px) per tap
// ============================================================
__global__ void offsets_kernel(const float* __restrict__ x,
                               const float* __restrict__ w_rot,
                               const float* __restrict__ b_rot,
                               const float* __restrict__ w_str,
                               const float* __restrict__ b_str,
                               const float* __restrict__ w_whole,
                               const float* __restrict__ b_whole,
                               float* __restrict__ offs) {
    __shared__ __align__(16) float wsm[4 * KDIM];
    int tid = threadIdx.x;
    for (int i = tid; i < 4 * KDIM; i += blockDim.x) {
        int f = i / KDIM, rem = i - f * KDIM;
        float v;
        if (f < 2)       v = w_rot[i];
        else if (f == 2) v = w_str[rem];
        else             v = w_whole[rem];
        wsm[i] = v;
    }
    __syncthreads();

    int p = blockIdx.x * blockDim.x + tid;
    if (p >= NPIX) return;
    int b  = p / HW;
    int hw = p - b * HW;
    int ph = hw / W;
    int pw = hw - ph * W;

    float s_sin = b_rot[0], s_cos = b_rot[1];
    float s_str = b_str[0], s_who = b_whole[0];

    const float* xb = x + ((long)b * CIN) * HW;
    for (int c = 0; c < CIN; ++c) {
        const float* xc = xb + (long)c * HW;
        const float* wc0 = wsm + c * KTAPS;
        #pragma unroll
        for (int t = 0; t < KTAPS; ++t) {
            int dy = t / 3 - 1, dx = t % 3 - 1;
            int yy = ph + dy, xx = pw + dx;
            float v = 0.f;
            if (yy >= 0 && yy < H && xx >= 0 && xx < W)
                v = xc[yy * W + xx];
            s_sin = fmaf(v, wc0[t],            s_sin);
            s_cos = fmaf(v, wc0[KDIM + t],     s_cos);
            s_str = fmaf(v, wc0[2 * KDIM + t], s_str);
            s_who = fmaf(v, wc0[3 * KDIM + t], s_who);
        }
    }

    float inv = rsqrtf(s_sin * s_sin + s_cos * s_cos + 1e-6f);
    float sn = s_sin * inv, cs = s_cos * inv;
    float r  = tanhf(s_str) * A_S + B_S;
    float wr = 1.0f + fmaxf(s_who, 0.0f);
    float rwr = r * wr;

    f32x2* out2 = (f32x2*)offs;
    #pragma unroll
    for (int t = 0; t < KTAPS; ++t) {
        float di = (float)(t / 3 - 1);
        float dj = (float)(t % 3 - 1);
        float bd0 = di * rwr;
        float bd1 = dj * wr;
        f32x2 o;
        o.x = (float)ph + cs * bd0 + sn * bd1;
        o.y = (float)pw - sn * bd0 + cs * bd1;
        out2[(long)p * KTAPS + t] = o;
    }
}

// ============================================================
// Kernel 4: implicit GEMM via v_wmma_f32_16x16x32_bf16
// WG = 256 thr = 8 waves; per wave M=32, N=64; WG tile = 256 px
// B-chunk staged per tap with async DMA (global_load_async_to_lds)
// ============================================================
__global__ void
__launch_bounds__(256, 1)
gemm_kernel(const float* __restrict__ xf,            // NHWC f32
            const float* __restrict__ offs,          // (py,px) per (p,tap)
            const unsigned short* __restrict__ wb,   // B-layout weights bf16
            float* __restrict__ out) {
    __shared__ __align__(16) unsigned short ldsB[2 * 4 * 16 * 32];   //  8 KB
    __shared__ __align__(16) unsigned short ldsA[8 * 32 * 64];       // 32 KB

    const int tid  = threadIdx.x;
    const int lane = tid & 31;
    const int wave = tid >> 5;
    const int px_i = lane >> 1;         // pixel 0..15 within half-tile
    const int half = lane & 1;          // channel half (32 ch each)
    const int row  = lane & 15;
    const int sel  = (lane >> 4) & 1;   // lanes 16..31 take the +16B K halves

    const long tileBase = (long)blockIdx.x * 256;    // 256 px per WG
    const long waveBase = tileBase + wave * 32;
    const int  bb       = (int)(tileBase / HW);      // whole tile in one batch
    const int  hwWave   = (int)(waveBase - (long)bb * HW);

    const v8f zero = {0.f,0.f,0.f,0.f,0.f,0.f,0.f,0.f};
    v8f acc[2][4];
    #pragma unroll
    for (int ms = 0; ms < 2; ++ms)
        #pragma unroll
        for (int nt = 0; nt < 4; ++nt) acc[ms][nt] = zero;

    unsigned short* aTile = ldsA + wave * (32 * 64);
    const f32x4* xf4  = (const f32x4*)xf;
    const f32x2* off2 = (const f32x2*)offs;

    // flat->LDS offset (low 32 bits of the shared aperture address)
    const unsigned ldsBOff = (unsigned)(unsigned long long)(void*)ldsB;

    for (int k = 0; k < KTAPS; ++k) {
        __syncthreads();    // WAR: previous tap finished reading LDS

        // ---- async DMA this tap's B chunk: 8KB = 512 x b128 ----
        {
            unsigned           dst = ldsBOff + (unsigned)tid * 16u;
            unsigned long long src = (unsigned long long)(wb + k * 4096)
                                     + (unsigned long long)tid * 16ull;
            asm volatile("global_load_async_to_lds_b128 %0, %1, off"
                         :: "v"(dst), "v"(src) : "memory");
            asm volatile("global_load_async_to_lds_b128 %0, %1, off offset:4096"
                         :: "v"(dst), "v"(src) : "memory");
            if (k + 1 < KTAPS)
                __builtin_prefetch(wb + (k + 1) * 4096, 0, 0);
        }

        // ---- bilinear gather (f32) -> bf16 A tiles in LDS ----
        #pragma unroll
        for (int sub = 0; sub < 2; ++sub) {
            const long p = waveBase + sub * 16 + px_i;
            f32x2 o = off2[p * KTAPS + k];
            float py = o.x, pxf = o.y;
            float yf = floorf(py), xff = floorf(pxf);
            float ty = py - yf, tx = pxf - xff;
            int y0 = (int)yf, x0 = (int)xff;
            int y1 = y0 + 1,  x1 = x0 + 1;
            float w00 = (1.f - ty) * (1.f - tx);
            float w01 = (1.f - ty) * tx;
            float w10 = ty * (1.f - tx);
            float w11 = ty * tx;
            w00 *= (y0 >= 0 && y0 < H && x0 >= 0 && x0 < W) ? 1.f : 0.f;
            w01 *= (y0 >= 0 && y0 < H && x1 >= 0 && x1 < W) ? 1.f : 0.f;
            w10 *= (y1 >= 0 && y1 < H && x0 >= 0 && x0 < W) ? 1.f : 0.f;
            w11 *= (y1 >= 0 && y1 < H && x1 >= 0 && x1 < W) ? 1.f : 0.f;
            int y0c = min(max(y0, 0), H - 1), y1c = min(max(y1, 0), H - 1);
            int x0c = min(max(x0, 0), W - 1), x1c = min(max(x1, 0), W - 1);
            long base = (long)bb * HW;
            long e00 = (((base + (long)y0c * W + x0c) * CIN) + half * 32) >> 2;
            long e01 = (((base + (long)y0c * W + x1c) * CIN) + half * 32) >> 2;
            long e10 = (((base + (long)y1c * W + x0c) * CIN) + half * 32) >> 2;
            long e11 = (((base + (long)y1c * W + x1c) * CIN) + half * 32) >> 2;

            u32x4* aDst = (u32x4*)(aTile + (sub * 16 + px_i) * 64 + half * 32);
            #pragma unroll
            for (int i = 0; i < 4; ++i) {            // 8 channels / iter
                f32x4 a0 = xf4[e00 + 2*i], a1 = xf4[e00 + 2*i + 1];
                f32x4 b0 = xf4[e01 + 2*i], b1 = xf4[e01 + 2*i + 1];
                f32x4 c0 = xf4[e10 + 2*i], c1 = xf4[e10 + 2*i + 1];
                f32x4 d0 = xf4[e11 + 2*i], d1 = xf4[e11 + 2*i + 1];
                u32x4 res;
                unsigned short* sr = (unsigned short*)&res;
                #pragma unroll
                for (int j = 0; j < 4; ++j) {
                    float v = fmaf(w00, a0[j],
                              fmaf(w01, b0[j],
                              fmaf(w10, c0[j], w11 * d0[j])));
                    sr[j] = f2bf(v);
                }
                #pragma unroll
                for (int j = 0; j < 4; ++j) {
                    float v = fmaf(w00, a1[j],
                              fmaf(w01, b1[j],
                              fmaf(w10, c1[j], w11 * d1[j])));
                    sr[4 + j] = f2bf(v);
                }
                aDst[i] = res;
            }
        }

        asm volatile("s_wait_asynccnt 0x0" ::: "memory");
        __syncthreads();

        // ---- WMMA: 2 K-blocks x (2 M-subtiles sharing each B frag) x 4 N ----
        #pragma unroll
        for (int kb = 0; kb < 2; ++kb) {
            v16bf af[2];
            #pragma unroll
            for (int ms = 0; ms < 2; ++ms) {
                const unsigned short* ap =
                    aTile + (ms * 16 + row) * 64 + kb * 32 + sel * 8;
                ((u32x4*)&af[ms])[0] = *(const u32x4*)(ap);
                ((u32x4*)&af[ms])[1] = *(const u32x4*)(ap + 16);
            }
            #pragma unroll
            for (int nt = 0; nt < 4; ++nt) {
                const unsigned short* bp =
                    ldsB + (((kb * 4 + nt) * 16 + row) * 32) + sel * 8;
                v16bf bf;
                ((u32x4*)&bf)[0] = *(const u32x4*)(bp);
                ((u32x4*)&bf)[1] = *(const u32x4*)(bp + 16);
                acc[0][nt] = __builtin_amdgcn_wmma_f32_16x16x32_bf16(
                                 false, af[0], false, bf, (short)0,
                                 acc[0][nt], false, false);
                acc[1][nt] = __builtin_amdgcn_wmma_f32_16x16x32_bf16(
                                 false, af[1], false, bf, (short)0,
                                 acc[1][nt], false, false);
            }
        }
    }

    // ---- store D as b128 runs (4 consecutive pixels per store) ----
    #pragma unroll
    for (int ms = 0; ms < 2; ++ms) {
        #pragma unroll
        for (int nt = 0; nt < 4; ++nt) {
            int o = nt * 16 + row;
            long base = ((long)(bb * COUT + o)) * HW
                        + hwWave + ms * 16 + sel * 8;
            f32x4 lo, hi;
            #pragma unroll
            for (int j = 0; j < 4; ++j) { lo[j] = acc[ms][nt][j];
                                          hi[j] = acc[ms][nt][4 + j]; }
            *(f32x4*)(out + base)     = lo;
            *(f32x4*)(out + base + 4) = hi;
        }
    }
}

// ============================================================
extern "C" void kernel_launch(void* const* d_in, const int* in_sizes, int n_in,
                              void* d_out, int out_size, void* d_ws, size_t ws_size,
                              hipStream_t stream) {
    const float* x       = (const float*)d_in[0];
    const float* w_main  = (const float*)d_in[1];
    const float* w_rot   = (const float*)d_in[2];
    const float* b_rot   = (const float*)d_in[3];
    const float* w_str   = (const float*)d_in[4];
    const float* b_str   = (const float*)d_in[5];
    const float* w_whole = (const float*)d_in[6];
    const float* b_whole = (const float*)d_in[7];
    float* out = (float*)d_out;

    char* ws = (char*)d_ws;
    float*          wsOff = (float*)(ws);
    float*          wsXf  = (float*)(ws + OFF_XF);
    unsigned short* wsWb  = (unsigned short*)(ws + OFF_WB);

    (void)in_sizes; (void)n_in; (void)out_size; (void)ws_size;

    prep_wmain_kernel<<<(KDIM * COUT + 255) / 256, 256, 0, stream>>>(w_main, wsWb);
    convert_x_kernel<<<(NPIX * CIN + 255) / 256, 256, 0, stream>>>(x, wsXf);
    offsets_kernel<<<NPIX / 256, 256, 0, stream>>>(x, w_rot, b_rot, w_str, b_str,
                                                   w_whole, b_whole, wsOff);
    gemm_kernel<<<NPIX / 256, 256, 0, stream>>>(wsXf, wsOff, wsWb, out);
}
